// MemAut_60043642798504
// MI455X (gfx1250) — matrix operand order repaired and checked by reading, not compiled, not run on hardware
//
#include <hip/hip_runtime.h>
#include <hip/hip_bf16.h>
#include <math.h>

typedef __attribute__((ext_vector_type(16))) __bf16 v16bf;
typedef __attribute__((ext_vector_type(8)))  float  v8f;

#define NUM_MEM 100
#define FEAT    576
#define WAVES   8
#define KT1     18           // 576/32 K-steps, GEMM1
#define NT1     7            // ceil(100/16) N-tiles, GEMM1
#define KT2     4            // 128/32 K-steps, GEMM2
#define NT2     36           // 576/16 N-tiles, GEMM2
#define FRAG    512          // elements per 16x32 bf16 B tile (32 lanes x 16)

// ---- LDS layout (bytes) ----
// B operands stored FRAGMENT-MAJOR: tile t, lane l, elem j at (t*512 + l*16 + j).
// Every inner-loop ds_load is then base_vgpr + small immediate offset.
#define OFF_MNSF  0
#define SZ_MNSF   (KT1 * NT1 * FRAG * 2)   // 129024  GEMM1 B tiles, order [kt][nt]
#define OFF_MNFF  (OFF_MNSF + SZ_MNSF)
#define SZ_MNFF   (NT2 * KT2 * FRAG * 2)   // 147456   GEMM2 B tiles, order [nt2][kt2]
#define OFF_WBUF  (OFF_MNFF + SZ_MNFF)
#define SZ_WBUF   (WAVES * 16 * 128 * 2)   // 32768   per-wave weight staging [16][128] bf16
#define OFF_NORM  (OFF_WBUF + SZ_WBUF)
#define SZ_NORM   (112 * 4)
#define OFF_SCALE (OFF_NORM + SZ_NORM)
#define SZ_SCALE  (112 * 4)
#define LDS_TOTAL (OFF_SCALE + SZ_SCALE)   // 310144 <= 320KB WGP LDS

// ds_swizzle xor-mode lane exchange: no address VGPR, no width clamping.
template <int MASK>
__device__ __forceinline__ float swz_xor(float x) {
    int i = __builtin_amdgcn_ds_swizzle(__float_as_int(x), (MASK << 10) | 0x1f);
    return __int_as_float(i);
}
__device__ __forceinline__ float red_max16(float x) {
    x = fmaxf(x, swz_xor<1>(x));
    x = fmaxf(x, swz_xor<2>(x));
    x = fmaxf(x, swz_xor<4>(x));
    x = fmaxf(x, swz_xor<8>(x));
    return x;
}
__device__ __forceinline__ float red_sum16(float x) {
    x += swz_xor<1>(x);
    x += swz_xor<2>(x);
    x += swz_xor<4>(x);
    x += swz_xor<8>(x);
    return x;
}

// fragment load: 32 contiguous bytes at base + compile-time element offset
#define LOADB(dst, basep, immElems) do {                                      \
    ((uint4*)&(dst))[0] = ((const uint4*)((basep) + (immElems)))[0];          \
    ((uint4*)&(dst))[1] = ((const uint4*)((basep) + (immElems) + 8))[0];      \
} while (0)

__global__ void __launch_bounds__(256, 1)
memaut_kernel(const float* __restrict__ z, const float* __restrict__ mem,
              float* __restrict__ out, int b)
{
    extern __shared__ char smem[];
    __bf16* mnSf    = (__bf16*)(smem + OFF_MNSF);
    __bf16* mnFf    = (__bf16*)(smem + OFF_MNFF);
    __bf16* wb0     = (__bf16*)(smem + OFF_WBUF);
    float*  memnorm = (float*)(smem + OFF_NORM);
    float*  mscale  = (float*)(smem + OFF_SCALE);

    const int tid = threadIdx.x;

    // ---- stage 1: memory row norms (100 rows, cache-resident) ----
    if (tid < 112) {
        float nrm = 0.f, sc = 0.f;
        if (tid < NUM_MEM) {
            float s = 0.f;
            const float* mr = mem + tid * FEAT;
            for (int k = 0; k < FEAT; ++k) { float v = mr[k]; s += v * v; }
            nrm = sqrtf(s);
            sc  = 1.f / fmaxf(nrm, 1e-8f);
        }
        memnorm[tid] = nrm;
        mscale[tid]  = sc;
    }
    __syncthreads();

    // ---- stage 2: normalized memory -> LDS, fragment-major (B-tile layout) ----
    // GEMM1 B: tile (kt,nt), lane l, elem j  ->  mn[n = nt*16 + l%16][k = kt*32 + (l/16)*8 + j + (j&8)]
    for (int e = tid; e < KT1 * NT1 * FRAG; e += 256) {
        int kt  = e / (NT1 * FRAG);
        int rem = e - kt * (NT1 * FRAG);
        int nt  = rem >> 9;
        int l   = (rem >> 4) & 31;
        int j   = rem & 15;
        int n   = nt * 16 + (l & 15);
        int k   = kt * 32 + (l >> 4) * 8 + j + (j & 8);
        float v = (n < NUM_MEM) ? mem[n * FEAT + k] * mscale[n] : 0.f;
        mnSf[e] = (__bf16)v;
    }
    // GEMM2 B: tile (nt2,kt2), lane l, elem j -> mn[slot = kt2*32 + (l/16)*8 + j + (j&8)][feat = nt2*16 + l%16]
    for (int e = tid; e < NT2 * KT2 * FRAG; e += 256) {
        int nt2  = e >> 11;
        int kt2  = (e >> 9) & 3;
        int l    = (e >> 4) & 31;
        int j    = e & 15;
        int slot = kt2 * 32 + (l >> 4) * 8 + j + (j & 8);
        int feat = nt2 * 16 + (l & 15);
        float v  = (slot < NUM_MEM) ? mem[slot * FEAT + feat] * mscale[slot] : 0.f;
        mnFf[e] = (__bf16)v;
    }
    __syncthreads();

    const int wave = tid >> 5;
    const int lane = tid & 31;
    const int h    = lane >> 4;   // lane half (WMMA fragment layout)
    const int m    = lane & 15;

    const long tile = (long)blockIdx.x * WAVES + wave;
    const long row0 = tile * 16;

    long rowA = row0 + m; if (rowA >= b) rowA = b - 1;
    const float* zr = z + rowA * FEAT;

    // ---- pass A: z row norms (lane covers half the row; xor-16 completes it) ----
    float sumsq = 0.f;
    for (int kt = 0; kt < KT1; ++kt) {
        const float4* p0 = (const float4*)(zr + kt * 32 + h * 8);
        const float4* p1 = (const float4*)(zr + kt * 32 + 16 + h * 8);
        float4 a0 = p0[0], a1 = p0[1], a2 = p1[0], a3 = p1[1];
        sumsq += a0.x*a0.x + a0.y*a0.y + a0.z*a0.z + a0.w*a0.w
               + a1.x*a1.x + a1.y*a1.y + a1.z*a1.z + a1.w*a1.w
               + a2.x*a2.x + a2.y*a2.y + a2.z*a2.z + a2.w*a2.w
               + a3.x*a3.x + a3.y*a3.y + a3.z*a3.z + a3.w*a3.w;
    }
    float total  = sumsq + swz_xor<16>(sumsq);
    float zscale = 1.f / fmaxf(sqrtf(total), 1e-8f);

    v8f zero8;
    #pragma unroll
    for (int i = 0; i < 8; ++i) zero8[i] = 0.f;

    // ---- GEMM1: logits[16x112] = zn[16x576] @ B tiles (pipelined, imm offsets) ----
    v8f acc[NT1];
    #pragma unroll
    for (int i = 0; i < NT1; ++i) acc[i] = zero8;

    const __bf16* pkt = mnSf + lane * 16;   // per-kt fragment base, +3584 elems per kt
    v16bf bcur, bnext;
    LOADB(bcur, pkt, 0);
    for (int kt = 0; kt < KT1; ++kt) {
        const float4* p0 = (const float4*)(zr + kt * 32 + h * 8);
        const float4* p1 = (const float4*)(zr + kt * 32 + 16 + h * 8);
        float4 a0 = p0[0], a1 = p0[1], a2 = p1[0], a3 = p1[1];
        v16bf afrag;
        afrag[0]  = (__bf16)(a0.x * zscale); afrag[1]  = (__bf16)(a0.y * zscale);
        afrag[2]  = (__bf16)(a0.z * zscale); afrag[3]  = (__bf16)(a0.w * zscale);
        afrag[4]  = (__bf16)(a1.x * zscale); afrag[5]  = (__bf16)(a1.y * zscale);
        afrag[6]  = (__bf16)(a1.z * zscale); afrag[7]  = (__bf16)(a1.w * zscale);
        afrag[8]  = (__bf16)(a2.x * zscale); afrag[9]  = (__bf16)(a2.y * zscale);
        afrag[10] = (__bf16)(a2.z * zscale); afrag[11] = (__bf16)(a2.w * zscale);
        afrag[12] = (__bf16)(a3.x * zscale); afrag[13] = (__bf16)(a3.y * zscale);
        afrag[14] = (__bf16)(a3.z * zscale); afrag[15] = (__bf16)(a3.w * zscale);

        #pragma unroll
        for (int nt = 0; nt < NT1; ++nt) {
            // prefetch next fragment: (kt, nt+1) or (kt+1, 0); all offsets are immediates
            LOADB(bnext, pkt, (nt + 1) * FRAG);
            acc[nt] = __builtin_amdgcn_wmma_f32_16x16x32_bf16(
                false, afrag, false, bcur, (short)0, acc[nt], false, false);
            bcur = bnext;
        }
        pkt += NT1 * FRAG;   // one pointer bump per K-row
    }

    // ---- softmax + hard-shrink + L1 renorm, in C-fragment layout ----
    const float TH = 1.0f / NUM_MEM;
    #pragma unroll
    for (int r = 0; r < 8; ++r) {
        float mx = -3.4e38f;
        #pragma unroll
        for (int nt = 0; nt < NT1; ++nt) {
            int N = m + 16 * nt;
            float v = acc[nt][r];
            mx = (N < NUM_MEM) ? fmaxf(mx, v) : mx;
        }
        mx = red_max16(mx);

        float s = 0.f;
        #pragma unroll
        for (int nt = 0; nt < NT1; ++nt) {
            int N = m + 16 * nt;
            float e = (N < NUM_MEM) ? __expf(acc[nt][r] - mx) : 0.f;
            acc[nt][r] = e; s += e;
        }
        s = red_sum16(s);
        float inv = 1.f / s;

        float s1 = 0.f;
        #pragma unroll
        for (int nt = 0; nt < NT1; ++nt) {
            float w = acc[nt][r] * inv;
            float d = w - TH;
            w = fmaxf(d, 0.f) * w / (fabsf(d) + 1e-15f);
            acc[nt][r] = w; s1 += w;
        }
        s1 = red_sum16(s1);
        float invs1 = 1.f / s1;

        #pragma unroll
        for (int nt = 0; nt < NT1; ++nt) {
            int N = m + 16 * nt;
            // fold memory row norm in: w @ memory == (w * ||m||) @ mn
            acc[nt][r] = acc[nt][r] * invs1 * memnorm[N];
        }
    }

    // ---- transpose weights to A-fragment layout via per-wave LDS staging ----
    __bf16* wb = wb0 + wave * 16 * 128;
    #pragma unroll
    for (int r = 0; r < 8; ++r) {
        int M = r + 8 * h;
        #pragma unroll
        for (int nt = 0; nt < NT1; ++nt)
            wb[M * 128 + m + 16 * nt] = (__bf16)acc[nt][r];
        wb[M * 128 + 112 + m] = (__bf16)0.f;   // zero K padding 112..127
    }
    asm volatile("s_wait_dscnt 0" ::: "memory"); // same-wave LDS RAW

    v16bf w4[KT2];
    #pragma unroll
    for (int kt = 0; kt < KT2; ++kt) {
        const __bf16* ap = wb + m * 128 + kt * 32 + h * 8;
        ((uint4*)&w4[kt])[0] = ((const uint4*)ap)[0];        // K run 0..7
        ((uint4*)&w4[kt])[1] = ((const uint4*)(ap + 16))[0]; // K run 16..23
    }

    // ---- GEMM2: z_hat[16x576] = w[16x128] @ B tiles ----
    // 2 N-tiles (independent acc chains) + one-step-ahead prefetch, imm offsets only.
    const __bf16* pb = mnFf + lane * 16;    // per-pair base, +4096 elems per pair
    v16bf ba_c, bb_c, ba_n, bb_n;
    LOADB(ba_c, pb, 0);
    LOADB(bb_c, pb, 4 * FRAG);
    for (int nt2 = 0; nt2 < NT2; nt2 += 2) {
        v8f acc2a = zero8, acc2b = zero8;
        #pragma unroll
        for (int kt = 0; kt < KT2; ++kt) {
            // next step: (pair, kt+1); at kt=3 offsets 4*FRAG / 8*FRAG
            LOADB(ba_n, pb, (kt + 1) * FRAG);
            LOADB(bb_n, pb, (kt + 5) * FRAG);
            acc2a = __builtin_amdgcn_wmma_f32_16x16x32_bf16(
                false, w4[kt], false, ba_c, (short)0, acc2a, false, false);
            acc2b = __builtin_amdgcn_wmma_f32_16x16x32_bf16(
                false, w4[kt], false, bb_c, (short)0, acc2b, false, false);
            ba_c = ba_n; bb_c = bb_n;
        }
        #pragma unroll
        for (int r = 0; r < 8; ++r) {
            long row = row0 + r + 8 * h;
            if (row < b) {
                out[row * FEAT + 16 * nt2 + m]      = acc2a[r];
                out[row * FEAT + 16 * nt2 + 16 + m] = acc2b[r];
            }
        }
        pb += 2 * KT2 * FRAG;   // one pointer bump per N-pair
    }
}

extern "C" void kernel_launch(void* const* d_in, const int* in_sizes, int n_in,
                              void* d_out, int out_size, void* d_ws, size_t ws_size,
                              hipStream_t stream) {
    (void)n_in; (void)out_size; (void)d_ws; (void)ws_size;
    const float* z   = (const float*)d_in[0];
    const float* mem = (const float*)d_in[1];
    float* out = (float*)d_out;

    int b      = in_sizes[0] / FEAT;
    int tiles  = (b + 15) / 16;
    int blocks = (tiles + WAVES - 1) / WAVES;

    hipFuncSetAttribute((const void*)memaut_kernel,
                        hipFuncAttributeMaxDynamicSharedMemorySize, LDS_TOTAL);
    memaut_kernel<<<blocks, 256, LDS_TOTAL, stream>>>(z, mem, out, b);
}
